// EventTransformerConv_12335146074171
// MI455X (gfx1250) — matrix operand order repaired and checked by reading, not compiled, not run on hardware
//
#include <hip/hip_runtime.h>
#include <hip/hip_bf16.h>
#include <math.h>

typedef __attribute__((ext_vector_type(16))) _Float16 v16h;
typedef __attribute__((ext_vector_type(8)))  _Float16 v8h;
typedef __attribute__((ext_vector_type(8)))  float    v8f;

#define NN   20000
#define EE   640000
#define GG   64
#define HH   128
#define FIN0 32
#define ED   16
#define CC   8

// ---------------------------------------------------------------------------
// utility kernels
// ---------------------------------------------------------------------------
__global__ void k_cvt_f16(const float* __restrict__ s, _Float16* __restrict__ d, int n) {
  int i = blockIdx.x * blockDim.x + threadIdx.x;
  int stride = gridDim.x * blockDim.x;
  for (; i < n; i += stride) d[i] = (_Float16)s[i];
}

// Pack weight matrix W[fin x 128] (row-major, f32) into the per-lane B-fragment
// order for v_wmma_f32_16x16x32_f16:
//   dest index = ((kcI*8 + t)*32 + lane)*16 + i
//   col = t*16 + (lane&15) ;  k = kcI*32 + (lane>>4)*16 + i
__global__ void k_pack_w(const float* __restrict__ w, _Float16* __restrict__ wp, int fin) {
  int n = fin * HH;
  int idx = blockIdx.x * blockDim.x + threadIdx.x;
  int stride = gridDim.x * blockDim.x;
  for (; idx < n; idx += stride) {
    int i    = idx & 15;
    int lane = (idx >> 4) & 31;
    int t    = (idx >> 9) & 7;
    int kcI  = idx >> 12;
    int col  = t * 16 + (lane & 15);
    int kk   = kcI * 32 + (lane >> 4) * 16 + i;
    wp[idx] = (_Float16)w[kk * HH + col];
  }
}

__global__ void k_init_nodes(float* __restrict__ m, float* __restrict__ den, int n) {
  int i = blockIdx.x * blockDim.x + threadIdx.x;
  if (i < n) { m[i] = __int_as_float(0xff800000); den[i] = 0.0f; }
}

__global__ void k_relu(float* __restrict__ h, int n) {
  int i = blockIdx.x * blockDim.x + threadIdx.x;
  if (i < n) h[i] = fmaxf(h[i], 0.0f);
}

// ---------------------------------------------------------------------------
// WMMA GEMM: out[nrows x 128] = A16[nrows x fin] @ Wpacked + bias
// one wave computes one 16-row tile x all 128 output columns
// ---------------------------------------------------------------------------
__global__ __launch_bounds__(128) void k_gemm_bias(
    const _Float16* __restrict__ A, const _Float16* __restrict__ Bp,
    const float* __restrict__ bias, float* __restrict__ out, int nrows, int fin) {
  int wave = blockIdx.x * (blockDim.x >> 5) + (threadIdx.x >> 5);
  int lane = threadIdx.x & 31;
  int ntiles = nrows >> 4;
  if (wave >= ntiles) return;          // wave-uniform: EXEC stays all-ones
  int m0 = wave << 4;
  int hl = lane & 15;
  int hi = lane >> 4;

  v8f acc[8] = {};
  int kch = fin >> 5;
  const _Float16* arow = A + (size_t)(m0 + hl) * fin;

  for (int kc = 0; kc < kch; ++kc) {
    int kb = kc * 32;
    // A-fragment (contiguous k, matches packed B): lanes 0-15: K0-15 ; lanes 16-31: K16-31
    v8h alo = *(const v8h*)(arow + kb + hi * 16);
    v8h ahi = *(const v8h*)(arow + kb + hi * 16 + 8);
    v16h a;
#pragma unroll
    for (int i = 0; i < 8; ++i) { a[i] = alo[i]; a[i + 8] = ahi[i]; }
#pragma unroll
    for (int t = 0; t < 8; ++t) {
      v16h b = *(const v16h*)(Bp + ((((size_t)kc * 8 + t) * 32 + lane) << 4));
      acc[t] = __builtin_amdgcn_wmma_f32_16x16x32_f16(
          false, a, false, b, (short)0, acc[t], false, false);
    }
  }
  // C/D layout: VGPR r -> row m0 + r + (hi?8:0), col = t*16 + hl
#pragma unroll
  for (int t = 0; t < 8; ++t) {
    int col = t * 16 + hl;
    float bv = bias[col];
    float* op = out + (size_t)(m0 + hi * 8) * HH + col;
#pragma unroll
    for (int r = 0; r < 8; ++r) op[(size_t)r * HH] = acc[t][r] + bv;
  }
}

// ---------------------------------------------------------------------------
// edge pipeline (wave per edge; lane owns 4 of 128 features)
// ---------------------------------------------------------------------------
__device__ inline void atomicMaxFloat(float* addr, float val) {
  if (val >= 0.0f) atomicMax((int*)addr, __float_as_int(val));
  else             atomicMin((unsigned int*)addr, __float_as_uint(val));
}

__global__ __launch_bounds__(256) void k_edge_alpha(
    const float* __restrict__ q, const float* __restrict__ k,
    const float* __restrict__ ea, const int* __restrict__ src,
    const int* __restrict__ dst, const float* __restrict__ we,
    const float* __restrict__ be, float* __restrict__ alpha,
    float* __restrict__ mbuf, int E) {
  __shared__ float we_s[ED * HH];
  __shared__ float be_s[HH];
  for (int i = threadIdx.x; i < ED * HH; i += blockDim.x) we_s[i] = we[i];
  for (int i = threadIdx.x; i < HH; i += blockDim.x) be_s[i] = be[i];
  __syncthreads();

  int wid  = blockIdx.x * (blockDim.x >> 5) + (threadIdx.x >> 5);
  int lane = threadIdx.x & 31;
  if (wid >= E) return;
  int s = src[wid], d = dst[wid];

  float ear[ED];
  const float4* eap = (const float4*)(ea + (size_t)wid * ED);
#pragma unroll
  for (int j = 0; j < 4; ++j) {
    float4 t = eap[j];
    ear[4 * j] = t.x; ear[4 * j + 1] = t.y; ear[4 * j + 2] = t.z; ear[4 * j + 3] = t.w;
  }
  int f0 = lane * 4;
  float4 qv = *(const float4*)(q + (size_t)d * HH + f0);
  float4 kv = *(const float4*)(k + (size_t)s * HH + f0);
  float qa[4] = {qv.x, qv.y, qv.z, qv.w};
  float ka[4] = {kv.x, kv.y, kv.z, kv.w};

  float partial = 0.0f;
#pragma unroll
  for (int c = 0; c < 4; ++c) {
    int f = f0 + c;
    float e = be_s[f];
#pragma unroll
    for (int j = 0; j < ED; ++j) e += ear[j] * we_s[j * HH + f];
    partial += qa[c] * (ka[c] + e);
  }
#pragma unroll
  for (int off = 16; off > 0; off >>= 1) partial += __shfl_xor(partial, off, 32);

  if (lane == 0) {
    float al = partial * 0.08838834764831845f;  // 1/sqrt(128)
    alpha[wid] = al;
    atomicMaxFloat(&mbuf[d], al);
  }
}

__global__ void k_edge_expsum(float* __restrict__ alpha, const float* __restrict__ m,
                              float* __restrict__ den, const int* __restrict__ dst, int E) {
  int i = blockIdx.x * blockDim.x + threadIdx.x;
  if (i >= E) return;
  int d = dst[i];
  float ex = expf(alpha[i] - m[d]);
  alpha[i] = ex;
  atomicAdd(&den[d], ex);
}

__global__ __launch_bounds__(256) void k_edge_msg(
    const float* __restrict__ v, const float* __restrict__ ea,
    const int* __restrict__ src, const int* __restrict__ dst,
    const float* __restrict__ we, const float* __restrict__ be,
    const float* __restrict__ exb, const float* __restrict__ den,
    float* __restrict__ out, int E) {
  __shared__ float we_s[ED * HH];
  __shared__ float be_s[HH];
  for (int i = threadIdx.x; i < ED * HH; i += blockDim.x) we_s[i] = we[i];
  for (int i = threadIdx.x; i < HH; i += blockDim.x) be_s[i] = be[i];
  __syncthreads();

  int wid  = blockIdx.x * (blockDim.x >> 5) + (threadIdx.x >> 5);
  int lane = threadIdx.x & 31;
  if (wid >= E) return;
  int s = src[wid], d = dst[wid];

  float ear[ED];
  const float4* eap = (const float4*)(ea + (size_t)wid * ED);
#pragma unroll
  for (int j = 0; j < 4; ++j) {
    float4 t = eap[j];
    ear[4 * j] = t.x; ear[4 * j + 1] = t.y; ear[4 * j + 2] = t.z; ear[4 * j + 3] = t.w;
  }
  float a = exb[wid] / den[d];
  int f0 = lane * 4;
  float4 vv = *(const float4*)(v + (size_t)s * HH + f0);
  float va[4] = {vv.x, vv.y, vv.z, vv.w};
#pragma unroll
  for (int c = 0; c < 4; ++c) {
    int f = f0 + c;
    float e = be_s[f];
#pragma unroll
    for (int j = 0; j < ED; ++j) e += ear[j] * we_s[j * HH + f];
    atomicAdd(&out[(size_t)d * HH + f], (va[c] + e) * a);
  }
}

// ---------------------------------------------------------------------------
// pooling + head
// ---------------------------------------------------------------------------
__global__ void k_pool_zero(float* __restrict__ sums, float* __restrict__ cnt) {
  int i = blockIdx.x * blockDim.x + threadIdx.x;
  if (i < GG * HH) sums[i] = 0.0f;
  if (i < GG) cnt[i] = 0.0f;
}

__global__ void k_pool_accum(const float* __restrict__ h, const int* __restrict__ batch,
                             float* __restrict__ sums, float* __restrict__ cnt) {
  int idx = blockIdx.x * blockDim.x + threadIdx.x;
  if (idx >= NN * HH) return;
  int n = idx >> 7, f = idx & 127;
  int g = batch[n];
  atomicAdd(&sums[(size_t)g * HH + f], h[idx]);
  if (f == 0) atomicAdd(&cnt[g], 1.0f);
}

__global__ void k_final(const float* __restrict__ sums, const float* __restrict__ cnt,
                        const float* __restrict__ lw, const float* __restrict__ lb,
                        float* __restrict__ out) {
  int tid = threadIdx.x;
  if (tid >= GG * CC) return;
  int g = tid >> 3, c = tid & 7;
  float cn = fmaxf(cnt[g], 1.0f);
  float acc = lb[c];
  for (int hf = 0; hf < HH; ++hf) acc += (sums[g * HH + hf] / cn) * lw[hf * CC + c];
  out[tid] = acc;
}

// ---------------------------------------------------------------------------
// launch
// ---------------------------------------------------------------------------
extern "C" void kernel_launch(void* const* d_in, const int* in_sizes, int n_in,
                              void* d_out, int out_size, void* d_ws, size_t ws_size,
                              hipStream_t stream) {
  (void)in_sizes; (void)n_in; (void)out_size; (void)ws_size;
  const float* x     = (const float*)d_in[0];
  const int*   ei    = (const int*)d_in[1];
  const float* ea    = (const float*)d_in[2];
  const int*   batch = (const int*)d_in[3];
  const int* src = ei;        // edge_index[0]
  const int* dst = ei + EE;   // edge_index[1]
  auto P = [&](int i) -> const float* { return (const float*)d_in[i]; };
  // params: conv layer base = 4 + 10*l ; +0 wq +1 bq +2 wk +3 bk +4 wv +5 bv
  //         +6 we +7 be +8 ws +9 bs ; lin_w=34 lin_b=35

  char* base = (char*)d_ws;
  size_t off = 0;
  auto takeB = [&](size_t bytes) -> char* {
    off = (off + 255) & ~(size_t)255;
    char* p = base + off; off += bytes; return p;
  };
  float* hA    = (float*)takeB((size_t)NN * HH * 4);
  float* hB    = (float*)takeB((size_t)NN * HH * 4);
  float* qb    = (float*)takeB((size_t)NN * HH * 4);
  float* kb    = (float*)takeB((size_t)NN * HH * 4);
  float* vb    = (float*)takeB((size_t)NN * HH * 4);
  float* mb    = (float*)takeB((size_t)NN * 4);
  float* den   = (float*)takeB((size_t)NN * 4);
  float* alpha = (float*)takeB((size_t)EE * 4);
  float* sums  = (float*)takeB((size_t)GG * HH * 4);
  float* cnt   = (float*)takeB((size_t)GG * 4);
  _Float16* xh  = (_Float16*)takeB((size_t)NN * HH * 2);
  _Float16* wqp = (_Float16*)takeB((size_t)HH * HH * 2);
  _Float16* wkp = (_Float16*)takeB((size_t)HH * HH * 2);
  _Float16* wvp = (_Float16*)takeB((size_t)HH * HH * 2);
  _Float16* wsp = (_Float16*)takeB((size_t)HH * HH * 2);

  auto run_layer = [&](const float* hin, int fin, int pb, float* hout, bool relu) {
    int nelem = NN * fin;
    k_cvt_f16<<<(nelem + 255) / 256, 256, 0, stream>>>(hin, xh, nelem);
    int wn = fin * HH;
    k_pack_w<<<(wn + 255) / 256, 256, 0, stream>>>(P(pb + 0), wqp, fin);
    k_pack_w<<<(wn + 255) / 256, 256, 0, stream>>>(P(pb + 2), wkp, fin);
    k_pack_w<<<(wn + 255) / 256, 256, 0, stream>>>(P(pb + 4), wvp, fin);
    k_pack_w<<<(wn + 255) / 256, 256, 0, stream>>>(P(pb + 8), wsp, fin);
    int gblocks = ((NN >> 4) + 3) / 4;  // 4 waves per block
    k_gemm_bias<<<gblocks, 128, 0, stream>>>(xh, wqp, P(pb + 1), qb, NN, fin);
    k_gemm_bias<<<gblocks, 128, 0, stream>>>(xh, wkp, P(pb + 3), kb, NN, fin);
    k_gemm_bias<<<gblocks, 128, 0, stream>>>(xh, wvp, P(pb + 5), vb, NN, fin);
    k_gemm_bias<<<gblocks, 128, 0, stream>>>(xh, wsp, P(pb + 9), hout, NN, fin); // skip
    k_init_nodes<<<(NN + 255) / 256, 256, 0, stream>>>(mb, den, NN);
    int eblocks = (EE + 7) / 8;  // 8 waves (edges) per 256-thread block
    k_edge_alpha<<<eblocks, 256, 0, stream>>>(qb, kb, ea, src, dst, P(pb + 6), P(pb + 7),
                                              alpha, mb, EE);
    k_edge_expsum<<<(EE + 255) / 256, 256, 0, stream>>>(alpha, mb, den, dst, EE);
    k_edge_msg<<<eblocks, 256, 0, stream>>>(vb, ea, src, dst, P(pb + 6), P(pb + 7),
                                            alpha, den, hout, EE);
    if (relu) k_relu<<<((NN * HH) + 255) / 256, 256, 0, stream>>>(hout, NN * HH);
  };

  run_layer(x,  FIN0, 4,  hA, true);
  run_layer(hA, HH,   14, hB, true);
  run_layer(hB, HH,   24, hA, false);

  k_pool_zero<<<(GG * HH + 255) / 256, 256, 0, stream>>>(sums, cnt);
  k_pool_accum<<<((NN * HH) + 255) / 256, 256, 0, stream>>>(hA, batch, sums, cnt);
  k_final<<<1, 512, 0, stream>>>(sums, cnt, P(34), P(35), (float*)d_out);
}
